// MultiHeadSelfAttention_30837865185684
// MI455X (gfx1250) — compile-verified, matrix operand order checked by reading, not run on hardware
//
#include <hip/hip_runtime.h>
#include <hip/hip_bf16.h>

// ---------------------------------------------------------------------------
// MI455X (gfx1250) multi-head self-attention, f16 WMMA path, software-pipelined.
//  - All GEMMs: v_wmma_f32_16x16x32_f16 (f32 accumulate).
//  - GEMM K-loop: manual 2-stage software pipeline in a DYNAMIC loop
//    (#pragma unroll 1).  Loads write directly into the alternate buffer set
//    (no rotation copies), and the dynamic loop keeps the scheduler from
//    sinking prefetch loads onto their WMMAs -> partial s_wait_loadcnt.
//  - Attention: flash online softmax in exp2 domain (0.125*log2e folded into
//    Q); K-frags double-buffered across iterations; wave-private LDS
//    transpose with wave_barrier only (same-wave DS ops are in-order).
//  - Fragment layouts per CDNA5 ISA 7.12.2 (wave32): A/B lane L holds
//    row/col = L%16; elems 0..7 -> K = 8*(L/16)+0..7, elems 8..15 ->
//    K = 16+8*(L/16)+0..7. C/D: lane L, vgpr r = (M = r+8*(L/16), N = L%16).
// ---------------------------------------------------------------------------

typedef __attribute__((ext_vector_type(16))) _Float16 v16h;
typedef __attribute__((ext_vector_type(8)))  _Float16 v8h;
typedef __attribute__((ext_vector_type(8)))  float    v8f;

namespace cfg {
constexpr int B   = 4;
constexpr int N   = 2048;
constexpr int C   = 768;
constexpr int H   = 12;
constexpr int D   = 64;            // head dim
constexpr int M   = B * N;         // 8192 rows
constexpr int OC3 = 3 * C;         // 2304 fused QKV outputs
}

#define DEV_INLINE static __device__ __forceinline__

DEV_INLINE v16h load_frag(const _Float16* row, int k0, int hf) {
  const v8h lo = *reinterpret_cast<const v8h*>(row + k0 + 8 * hf);
  const v8h hi = *reinterpret_cast<const v8h*>(row + k0 + 16 + 8 * hf);
  v16h r;
#pragma unroll
  for (int i = 0; i < 8; ++i) { r[i] = lo[i]; r[i + 8] = hi[i]; }
  return r;
}

DEV_INLINE v8f wmma_f16(v16h a, v16h b, v8f c) {
  return __builtin_amdgcn_wmma_f32_16x16x32_f16(
      /*neg_a=*/false, a, /*neg_b=*/false, b,
      /*c_mod=*/(short)0, c, /*reuse_a=*/false, /*reuse_b=*/false);
}

// Reductions across the 16-lane half-group (masks 1,2,4,8 stay inside it).
DEV_INLINE float red_max16(float v) {
#pragma unroll
  for (int off = 1; off < 16; off <<= 1) v = fmaxf(v, __shfl_xor(v, off, 32));
  return v;
}
DEV_INLINE float red_sum16(float v) {
#pragma unroll
  for (int off = 1; off < 16; off <<= 1) v += __shfl_xor(v, off, 32);
  return v;
}

// ---------------------------------------------------------------------------
// Kernel 0: fp32 -> f16 conversion (x, w_qkv, w_out).
// ---------------------------------------------------------------------------
__global__ void cvt_f32_f16(const float* __restrict__ in,
                            _Float16* __restrict__ out, int n) {
  for (int i = blockIdx.x * blockDim.x + threadIdx.x; i < n;
       i += gridDim.x * blockDim.x)
    out[i] = (_Float16)in[i];
}

// ---------------------------------------------------------------------------
// Pipelined 16x64-tile GEMM core: acc[t] += A(16xK) * B[t](Kx16), K = C = 768.
// 24 K-steps; manual 2-deep pipeline, dynamic loop (11 iters x 2 steps),
// prologue step 0, epilogue steps 22-23.  No buffer-rotation copies.
// ---------------------------------------------------------------------------
DEV_INLINE void gemm_16x64(const _Float16* arow, const _Float16* const* brow,
                           int hf, v8f acc[4]) {
  v16h a0, a1, b0[4], b1[4];
  a0 = load_frag(arow, 0, hf);
#pragma unroll
  for (int t = 0; t < 4; ++t) b0[t] = load_frag(brow[t], 0, hf);

  int k = 32;
#pragma unroll 1
  for (int i = 0; i < 11; ++i) {
    // stage A: prefetch odd step into buf1, compute even step from buf0
    a1 = load_frag(arow, k, hf);
#pragma unroll
    for (int t = 0; t < 4; ++t) b1[t] = load_frag(brow[t], k, hf);
#pragma unroll
    for (int t = 0; t < 4; ++t) acc[t] = wmma_f16(a0, b0[t], acc[t]);
    // stage B: prefetch next even step into buf0, compute odd step from buf1
    a0 = load_frag(arow, k + 32, hf);
#pragma unroll
    for (int t = 0; t < 4; ++t) b0[t] = load_frag(brow[t], k + 32, hf);
#pragma unroll
    for (int t = 0; t < 4; ++t) acc[t] = wmma_f16(a1, b1[t], acc[t]);
    k += 64;
  }
  // Drain: buf0 holds step 22 (offset 704); load step 23 (offset 736).
  a1 = load_frag(arow, 736, hf);
#pragma unroll
  for (int t = 0; t < 4; ++t) b1[t] = load_frag(brow[t], 736, hf);
#pragma unroll
  for (int t = 0; t < 4; ++t) acc[t] = wmma_f16(a0, b0[t], acc[t]);
#pragma unroll
  for (int t = 0; t < 4; ++t) acc[t] = wmma_f16(a1, b1[t], acc[t]);
}

// ---------------------------------------------------------------------------
// Kernel 1: fused QKV projection.  qkv[m,o] = x[m,:].w_qkv[o,:] + b_qkv[o].
// Epilogue scatters Q (pre-scaled 0.125*log2e for exp2-domain softmax) and K
// as [B,H,N,D], V transposed as [B,H,D,N] so P.V B-frags are contiguous.
// ---------------------------------------------------------------------------
__global__ void __launch_bounds__(256) gemm_qkv_f16(
    const _Float16* __restrict__ xh, const _Float16* __restrict__ wh,
    const float* __restrict__ bias, _Float16* __restrict__ qh,
    _Float16* __restrict__ kh, _Float16* __restrict__ vth) {
  const int lane = threadIdx.x & 31;
  const int l16 = lane & 15, hf = lane >> 4;
  const int wid = blockIdx.x * (blockDim.x >> 5) + (threadIdx.x >> 5);
  const int nTiles = cfg::OC3 / 64;        // 36
  const int ntg = wid % nTiles;
  const int mt  = wid / nTiles;            // 0..511
  const int o_base = ntg * 64;

  const _Float16* arow = xh + (size_t)(mt * 16 + l16) * cfg::C;
  const _Float16* brow[4];
#pragma unroll
  for (int t = 0; t < 4; ++t)
    brow[t] = wh + (size_t)(o_base + t * 16 + l16) * cfg::C;

  v8f acc[4] = {};
  gemm_16x64(arow, brow, hf, acc);

  // 64-wide tile stays within one (q/k/v, head) slice: o_base % 64 == 0.
  const int s = o_base / cfg::C;                 // 0=q 1=k 2=v
  const int h = (o_base % cfg::C) / cfg::D;      // head
  const float qscale = 0.125f * 1.44269504088896340736f;  // 1/sqrt(D)*log2(e)
#pragma unroll
  for (int t = 0; t < 4; ++t) {
    const int o = o_base + t * 16 + l16;
    const float bv = bias[o];
    const int d = t * 16 + l16;                  // dim within head
#pragma unroll
    for (int r = 0; r < 8; ++r) {
      const int m = mt * 16 + r + 8 * hf;
      const int bb = m >> 11;                    // / 2048
      const int nn = m & 2047;
      const size_t bh = (size_t)bb * cfg::H + h;
      const float val = acc[t][r] + bv;
      if (s == 0)
        qh[(bh * cfg::N + nn) * cfg::D + d] = (_Float16)(val * qscale);
      else if (s == 1)
        kh[(bh * cfg::N + nn) * cfg::D + d] = (_Float16)val;
      else
        vth[(bh * cfg::D + d) * cfg::N + nn] = (_Float16)val;
    }
  }
}

// ---------------------------------------------------------------------------
// Kernel 2: flash attention.  One wave = one (b,h,16-query tile).
// Steady state per 32-key iteration:
//   V(kt) loads issued (in-order queue behind K(kt) from last iteration),
//   score WMMAs on K(kt) [already landed], K(kt+1) prefetch issued,
//   exp2-domain online softmax, wave-private LDS transpose, 4 P.V WMMAs.
// ---------------------------------------------------------------------------
__global__ void __launch_bounds__(128) attn_flash_f16(
    const _Float16* __restrict__ qh, const _Float16* __restrict__ kh,
    const _Float16* __restrict__ vth, _Float16* __restrict__ ctxh) {
  __shared__ _Float16 plds_all[4][16 * 32];     // 1KB per wave, wave-private
  const int wslot = threadIdx.x >> 5;
  const int lane = threadIdx.x & 31;
  const int l16 = lane & 15, hf = lane >> 4;
  const int wave = blockIdx.x * 4 + wslot;      // 0..6143
  const int qt = wave & 127;                    // query tile
  const int bh = wave >> 7;                     // (b*H + h)
  _Float16* plds = plds_all[wslot];

  const _Float16* qbase = qh + (size_t)bh * cfg::N * cfg::D;
  const _Float16* kbase = kh + (size_t)bh * cfg::N * cfg::D;
  const _Float16* vbase = vth + (size_t)bh * cfg::D * cfg::N;

  // Q fragments (K-dim 64 = 2 chunks of 32), resident all loop.
  const _Float16* qrow = qbase + (size_t)(qt * 16 + l16) * cfg::D;
  const v16h qf0 = load_frag(qrow, 0, hf);
  const v16h qf1 = load_frag(qrow, 32, hf);

  // V^T row pointers per d-subtile, resident all loop.
  const _Float16* vrow[4];
#pragma unroll
  for (int dt = 0; dt < 4; ++dt)
    vrow[dt] = vbase + (size_t)(dt * 16 + l16) * cfg::N;

  const _Float16* krowA = kbase + (size_t)l16 * cfg::D;        // keys l16
  const _Float16* krowB = kbase + (size_t)(16 + l16) * cfg::D; // keys 16+l16

  v8f O[4] = {};
  float mrow[8], lrow[8];
#pragma unroll
  for (int r = 0; r < 8; ++r) { mrow[r] = -1e30f; lrow[r] = 0.f; }

  constexpr int NT = cfg::N / 32;               // 64 key tiles

  // Prologue: K fragments for kt = 0.
  v16h kf[2][4];
  kf[0][0] = load_frag(krowA, 0, hf);
  kf[0][1] = load_frag(krowA, 32, hf);
  kf[0][2] = load_frag(krowB, 0, hf);
  kf[0][3] = load_frag(krowB, 32, hf);

#pragma unroll 2
  for (int kt = 0; kt < NT; ++kt) {
    const int cur = kt & 1, nxt = cur ^ 1;      // static under unroll 2
    const int key0 = kt * 32;

    // --- V loads for this iteration (queue behind current K loads) ---
    v16h vf[4];
#pragma unroll
    for (int dt = 0; dt < 4; ++dt) vf[dt] = load_frag(vrow[dt], key0, hf);

    // --- scores: S(16q x 32k) as two 16x16 C-frags (log2 domain) ---
    v8f s0 = {}, s1 = {};
    s0 = wmma_f16(qf0, kf[cur][0], s0);
    s0 = wmma_f16(qf1, kf[cur][1], s0);
    s1 = wmma_f16(qf0, kf[cur][2], s1);
    s1 = wmma_f16(qf1, kf[cur][3], s1);

    // --- prefetch next iteration's K frags (wrap avoids a tail branch) ---
    const size_t koffn = (size_t)(((kt + 1) & (NT - 1)) * 32) * cfg::D;
    kf[nxt][0] = load_frag(krowA + koffn, 0, hf);
    kf[nxt][1] = load_frag(krowA + koffn, 32, hf);
    kf[nxt][2] = load_frag(krowB + koffn, 0, hf);
    kf[nxt][3] = load_frag(krowB + koffn, 32, hf);

    // --- exp2-domain online softmax (V + K-next loads fly under this) ---
    float alpha[8];
#pragma unroll
    for (int r = 0; r < 8; ++r) {
      float mx = red_max16(fmaxf(s0[r], s1[r]));
      const float mn = fmaxf(mrow[r], mx);
      alpha[r] = exp2f(mrow[r] - mn);
      mrow[r] = mn;
      const float p0 = exp2f(s0[r] - mn);
      const float p1 = exp2f(s1[r] - mn);
      s0[r] = p0; s1[r] = p1;
      lrow[r] = lrow[r] * alpha[r] + red_sum16(p0 + p1);
    }

    // --- C-layout -> A-layout for P via wave-private LDS tile.
    // Same-wave DS ops execute in order (DScnt, ISA 7.3); wave_barrier is a
    // zero-cost compiler fence, no s_barrier needed.
    __builtin_amdgcn_wave_barrier();
#pragma unroll
    for (int r = 0; r < 8; ++r) {
      const int m = r + 8 * hf;
      plds[m * 32 + l16]      = (_Float16)s0[r];
      plds[m * 32 + 16 + l16] = (_Float16)s1[r];
    }
    __builtin_amdgcn_wave_barrier();
    const v16h pf = load_frag(plds + l16 * 32, 0, hf);

    // --- rescale O, accumulate P.V ---
#pragma unroll
    for (int dt = 0; dt < 4; ++dt) {
#pragma unroll
      for (int r = 0; r < 8; ++r) O[dt][r] *= alpha[r];
      O[dt] = wmma_f16(pf, vf[dt], O[dt]);
    }
  }

  // --- normalize and write ctx[b, n, h*64 + d] as f16 ---
  const int bb = bh / cfg::H, h = bh % cfg::H;
#pragma unroll
  for (int dt = 0; dt < 4; ++dt) {
#pragma unroll
    for (int r = 0; r < 8; ++r) {
      const int nq = qt * 16 + r + 8 * hf;
      const float val = O[dt][r] / lrow[r];
      ctxh[((size_t)bb * cfg::N + nq) * cfg::C + h * cfg::D + dt * 16 + l16] =
          (_Float16)val;
    }
  }
}

// ---------------------------------------------------------------------------
// Kernel 3: output projection.  out[m,o] = ctx[m,:].w_out[o,:] + b_out[o].
// ---------------------------------------------------------------------------
__global__ void __launch_bounds__(256) gemm_out_f16(
    const _Float16* __restrict__ ctxh, const _Float16* __restrict__ wh,
    const float* __restrict__ bias, float* __restrict__ out) {
  const int lane = threadIdx.x & 31;
  const int l16 = lane & 15, hf = lane >> 4;
  const int wid = blockIdx.x * (blockDim.x >> 5) + (threadIdx.x >> 5);
  const int nTiles = cfg::C / 64;              // 12
  const int ntg = wid % nTiles;
  const int mt  = wid / nTiles;                // 0..511
  const int o_base = ntg * 64;

  const _Float16* arow = ctxh + (size_t)(mt * 16 + l16) * cfg::C;
  const _Float16* brow[4];
#pragma unroll
  for (int t = 0; t < 4; ++t)
    brow[t] = wh + (size_t)(o_base + t * 16 + l16) * cfg::C;

  v8f acc[4] = {};
  gemm_16x64(arow, brow, hf, acc);

#pragma unroll
  for (int t = 0; t < 4; ++t) {
    const int o = o_base + t * 16 + l16;
    const float bv = bias[o];
#pragma unroll
    for (int r = 0; r < 8; ++r) {
      const int m = mt * 16 + r + 8 * hf;
      out[(size_t)m * cfg::C + o] = acc[t][r] + bv;
    }
  }
}

// ---------------------------------------------------------------------------
// Host launcher.
// ---------------------------------------------------------------------------
extern "C" void kernel_launch(void* const* d_in, const int* in_sizes, int n_in,
                              void* d_out, int out_size, void* d_ws,
                              size_t ws_size, hipStream_t stream) {
  const float* x     = (const float*)d_in[0];
  const float* w_qkv = (const float*)d_in[1];
  const float* b_qkv = (const float*)d_in[2];
  const float* w_out = (const float*)d_in[3];
  const float* b_out = (const float*)d_in[4];
  float* out = (float*)d_out;

  // Workspace layout (f16 elements).
  _Float16* ws = (_Float16*)d_ws;
  size_t off = 0;
  const size_t nXh   = (size_t)cfg::M * cfg::C;                    // 6.29M
  const size_t nWq   = (size_t)cfg::OC3 * cfg::C;                  // 1.77M
  const size_t nWo   = (size_t)cfg::C * cfg::C;                    // 0.59M
  const size_t nHead = (size_t)cfg::B * cfg::H * cfg::N * cfg::D;  // 6.29M
  _Float16* xh    = ws + off; off += nXh;
  _Float16* wqkvh = ws + off; off += nWq;
  _Float16* wouth = ws + off; off += nWo;
  _Float16* qh    = ws + off; off += nHead;
  _Float16* kh    = ws + off; off += nHead;
  _Float16* vth   = ws + off; off += nHead;
  _Float16* ctxh  = ws + off; off += nXh;
  (void)ws_size; (void)in_sizes; (void)n_in; (void)out_size;

  // 0) fp32 -> f16
  cvt_f32_f16<<<1024, 256, 0, stream>>>(x, xh, (int)nXh);
  cvt_f32_f16<<<1024, 256, 0, stream>>>(w_qkv, wqkvh, (int)nWq);
  cvt_f32_f16<<<1024, 256, 0, stream>>>(w_out, wouth, (int)nWo);

  // 1) QKV projection: 512 M-tiles * 36 N-tiles = 18432 waves / 8 per block.
  gemm_qkv_f16<<<2304, 256, 0, stream>>>(xh, wqkvh, b_qkv, qh, kh, vth);

  // 2) Flash attention: 4*12*128 = 6144 waves / 4 per block.
  attn_flash_f16<<<1536, 128, 0, stream>>>(qh, kh, vth, ctxh);

  // 3) Output projection: 512 * 12 = 6144 waves / 8 per block.
  gemm_out_f16<<<768, 256, 0, stream>>>(ctxh, wouth, b_out, out);
}
